// TGC_57870389346991
// MI455X (gfx1250) — matrix operand order repaired and checked by reading, not compiled, not run on hardware
//
#include <hip/hip_runtime.h>
#include <hip/hip_bf16.h>
#include <cstdint>

// ---------------- problem dims ----------------
#define B_   32
#define N_   1024
#define T_   32
#define F_   5
#define H_   64
#define R_   5
#define G4H  256      // 4*H

// LDS row paddings (bank-conflict avoidance: rows stride to odd multiples of 4 banks)
#define WHH_S 72      // bf16 elems per whh row   (144 B)
#define HB_S  72      // bf16 elems per hbuf row  (144 B)
#define GB_S  260     // f32 elems per gbuf row   (1040 B)
#define WB_S  36      // f32 elems per wbuf row   (144 B)
#define RED_S 68      // f32 elems per red row    (272 B)

// ---------------- WMMA types (CDNA5 wave32) ----------------
typedef __attribute__((ext_vector_type(16))) __bf16       v16bf;
typedef __attribute__((ext_vector_type(8)))  float        v8f;
typedef __attribute__((ext_vector_type(4)))  unsigned int uint4v;  // trivial 16B chunk

union BFrag { v16bf v; uint4v q[2]; };   // 8 VGPRs = 16 bf16

// fast transcendentals: v_rcp_f32 + v_exp_f32 (no IEEE div pipeline)
__device__ __forceinline__ float rcp_(float x) { return __builtin_amdgcn_rcpf(x); }
__device__ __forceinline__ float sigmoidf_(float x) { return rcp_(1.f + __expf(-x)); }
__device__ __forceinline__ float tanhf_(float x) {
  x = fminf(fmaxf(x, -15.f), 15.f);
  const float e = __expf(2.f * x);
  return (e - 1.f) * rcp_(e + 1.f);
}

// =====================================================================
// Kernel 1: fused LSTM (T=32 steps) + temporal attention -> e_old
// One block = 16 sequences, 256 threads = 8 waves.
// Whh resident as per-wave bf16 B-fragments across all timesteps; the
// x@Wih^T pre-activation is computed per-lane straight into the WMMA C
// fragment (no LDS round-trip, 2 barriers per step instead of 3).
// =====================================================================
struct LstmSmem {
  __bf16 whh[G4H * WHH_S];   // Whh rows padded (B operand source)
  float  wih[G4H * F_];
  float  bias[G4H];          // bih + bhh
  float  attnw[H_];
  float  xtile[16 * T_ * F_];
  float  gbuf[16 * GB_S];    // WMMA D (gates) rows padded
  float  cbuf[16 * H_];      // cell state, f32
  __bf16 hbuf[16 * HB_S];    // hidden, bf16, rows padded (A operand source)
  __bf16 hs[T_][16 * H_];    // all hiddens for attention
  float  scores[16 * 33];
};

__global__ void lstm_attn_kernel(const float* __restrict__ x,
                                 const float* __restrict__ Wih,
                                 const float* __restrict__ Whh,
                                 const float* __restrict__ bih,
                                 const float* __restrict__ bhh,
                                 const float* __restrict__ attn_w,
                                 const float* __restrict__ attn_b,
                                 float*  __restrict__ e_old,
                                 __bf16* __restrict__ e_bf,
                                 __bf16* __restrict__ eT_bf)
{
  extern __shared__ char smem_raw[];
  LstmSmem& S = *reinterpret_cast<LstmSmem*>(smem_raw);
  const int tid  = threadIdx.x;
  const int wave = tid >> 5;
  const int lane = tid & 31;
  const int g    = lane >> 4;     // lane half
  const int nl   = lane & 15;
  const int bn0  = blockIdx.x * 16;

  // --- stage weights / inputs into LDS ---
  for (int i = tid; i < G4H * H_; i += 256) {
    const int n = i >> 6, k = i & 63;
    S.whh[n * WHH_S + k] = (__bf16)Whh[i];
  }
  for (int i = tid; i < G4H * F_; i += 256) S.wih[i] = Wih[i];
  for (int i = tid; i < G4H;      i += 256) S.bias[i] = bih[i] + bhh[i];
  for (int i = tid; i < H_;       i += 256) S.attnw[i] = attn_w[i];
  for (int i = tid; i < 16 * T_ * F_; i += 256) {
    const int m = i / (T_ * F_); const int rem = i - m * (T_ * F_);
    S.xtile[i] = x[(size_t)(bn0 + m) * (T_ * F_) + rem];
  }
  for (int i = tid; i < 16 * H_;   i += 256) S.cbuf[i] = 0.f;
  for (int i = tid; i < 16 * HB_S; i += 256) S.hbuf[i] = (__bf16)0.f;
  __syncthreads();

  // --- per-wave resident B fragments of Whh: cols [wave*32, wave*32+32) ---
  // B 32x16 bf16: lane holds col n = base + nl, elems k = {g*8+e, 16+g*8+e} (+32 per k-chunk)
  BFrag bw[2][2];   // [n-subtile][k-chunk]
  #pragma unroll
  for (int s = 0; s < 2; ++s) {
    const int n = wave * 32 + s * 16 + nl;
    const uint4v* row = reinterpret_cast<const uint4v*>(&S.whh[n * WHH_S]);
    #pragma unroll
    for (int kc = 0; kc < 2; ++kc) {
      const int qi = (kc * 32 + g * 8) >> 3;
      bw[s][kc].q[0] = row[qi];
      bw[s][kc].q[1] = row[qi + 2];
    }
  }
  // --- per-lane Wih columns + bias for this lane's two output columns ---
  const int n0 = wave * 32 + nl, n1 = n0 + 16;
  float wv0[F_], wv1[F_];
  #pragma unroll
  for (int f = 0; f < F_; ++f) { wv0[f] = S.wih[n0 * F_ + f]; wv1[f] = S.wih[n1 * F_ + f]; }
  const float b0 = S.bias[n0], b1 = S.bias[n1];

  // ---------------- timestep loop ----------------
  for (int t = 0; t < T_; ++t) {
    // A fragments from hbuf: lane row m = nl; elems k = kc*32 + {g*8..,16+g*8..}
    BFrag fa[2];
    {
      const uint4v* hrow = reinterpret_cast<const uint4v*>(&S.hbuf[nl * HB_S]);
      #pragma unroll
      for (int kc = 0; kc < 2; ++kc) {
        fa[kc].q[0] = hrow[(kc * 32 +      g * 8) >> 3];
        fa[kc].q[1] = hrow[(kc * 32 + 16 + g * 8) >> 3];
      }
    }
    // C fragment = bias + x_t @ Wih^T computed per-lane in registers
    v8f acc0, acc1;
    #pragma unroll
    for (int v = 0; v < 8; ++v) {
      const float* xv = &S.xtile[(v + 8 * g) * (T_ * F_) + t * F_];
      const float x0 = xv[0], x1 = xv[1], x2 = xv[2], x3 = xv[3], x4 = xv[4];
      acc0[v] = b0 + x0*wv0[0] + x1*wv0[1] + x2*wv0[2] + x3*wv0[3] + x4*wv0[4];
      acc1[v] = b1 + x0*wv1[0] + x1*wv1[1] + x2*wv1[2] + x3*wv1[3] + x4*wv1[4];
    }
    // gates += h @ Whh^T
    acc0 = __builtin_amdgcn_wmma_f32_16x16x32_bf16(false, fa[0].v, false, bw[0][0].v,
                                                   (short)0, acc0, false, false);
    acc0 = __builtin_amdgcn_wmma_f32_16x16x32_bf16(false, fa[1].v, false, bw[0][1].v,
                                                   (short)0, acc0, false, false);
    acc1 = __builtin_amdgcn_wmma_f32_16x16x32_bf16(false, fa[0].v, false, bw[1][0].v,
                                                   (short)0, acc1, false, false);
    acc1 = __builtin_amdgcn_wmma_f32_16x16x32_bf16(false, fa[1].v, false, bw[1][1].v,
                                                   (short)0, acc1, false, false);
    #pragma unroll
    for (int v = 0; v < 8; ++v) {
      S.gbuf[(v + 8 * g) * GB_S + n0] = acc0[v];
      S.gbuf[(v + 8 * g) * GB_S + n1] = acc1[v];
    }
    __syncthreads();

    // gate nonlinearities (torch order i,f,g,o)
    for (int cell = tid; cell < 16 * H_; cell += 256) {
      const int m = cell >> 6, n = cell & 63;
      const float ig = S.gbuf[m * GB_S + n];
      const float fg = S.gbuf[m * GB_S + 64  + n];
      const float gg = S.gbuf[m * GB_S + 128 + n];
      const float og = S.gbuf[m * GB_S + 192 + n];
      const float c  = sigmoidf_(fg) * S.cbuf[cell] + sigmoidf_(ig) * tanhf_(gg);
      const float h  = sigmoidf_(og) * tanhf_(c);
      S.cbuf[cell] = c;
      const __bf16 hb = (__bf16)h;
      S.hbuf[m * HB_S + n] = hb;
      S.hs[t][cell]        = hb;
    }
    __syncthreads();
  }

  // ---------------- attention ----------------
  for (int s = tid; s < 16 * T_; s += 256) {
    const int m = s >> 5, tt = s & 31;
    float acc = attn_b[0];
    const __bf16* hp = &S.hs[tt][m * H_];
    #pragma unroll 8
    for (int k = 0; k < H_; ++k) acc += (float)hp[k] * S.attnw[k];
    S.scores[m * 33 + tt] = tanhf_(acc);
  }
  __syncthreads();
  if (tid < 16) {
    float mx = -1e30f;
    for (int tt = 0; tt < T_; ++tt) mx = fmaxf(mx, S.scores[tid * 33 + tt]);
    float sum = 0.f;
    for (int tt = 0; tt < T_; ++tt) {
      const float e = __expf(S.scores[tid * 33 + tt] - mx);
      S.scores[tid * 33 + tt] = e; sum += e;
    }
    const float inv = rcp_(sum);
    for (int tt = 0; tt < T_; ++tt) S.scores[tid * 33 + tt] *= inv;
  }
  __syncthreads();
  for (int cell = tid; cell < 16 * H_; cell += 256) {
    const int m = cell >> 6, h = cell & 63;
    float acc = 0.f;
    #pragma unroll 8
    for (int tt = 0; tt < T_; ++tt) acc += S.scores[m * 33 + tt] * (float)S.hs[tt][m * H_ + h];
    const int bn = bn0 + m;
    e_old[(size_t)bn * H_ + h] = acc;
    const __bf16 ab = (__bf16)acc;
    e_bf[(size_t)bn * H_ + h] = ab;
    const int b = bn >> 10, node = bn & 1023;
    eT_bf[((size_t)b * H_ + h) * N_ + node] = ab;   // E^T for 2nd-GEMM B operand
  }
}

// =====================================================================
// Kernel 2: coef[i][j] = mask * leaky_relu(A[i,j,:]@gw + gb), deg[i]
// =====================================================================
__global__ void coef_kernel(const float* __restrict__ A,
                            const float* __restrict__ gw, const float* __restrict__ gb,
                            float* __restrict__ coef, float* __restrict__ deg)
{
  __shared__ float red[256];
  const int i = blockIdx.x, tid = threadIdx.x;
  const float w0 = gw[0], w1 = gw[1], w2 = gw[2], w3 = gw[3], w4 = gw[4], b = gb[0];
  float cnt = 0.f;
  for (int j = tid; j < N_; j += 256) {
    const float* a = &A[((size_t)i * N_ + j) * R_];
    const float s = a[0] + a[1] + a[2] + a[3] + a[4];
    const float d = a[0]*w0 + a[1]*w1 + a[2]*w2 + a[3]*w3 + a[4]*w4 + b;
    const float lr = d > 0.f ? d : 0.2f * d;
    const float m  = s > 0.f ? 1.f : 0.f;
    coef[(size_t)i * N_ + j] = m * lr;
    cnt += m;
  }
  red[tid] = cnt;
  __syncthreads();
  for (int s = 128; s > 0; s >>= 1) {
    if (tid < s) red[tid] += red[tid + s];
    __syncthreads();
  }
  if (tid == 0) deg[i] = red[0];
}

// =====================================================================
// Kernel 3: fused GNN  e_new = ((E E^T) o coef) E / deg  — no dots tensor.
// One block = (batch b, 16-row i-tile); 8 waves split j; per 32-j chunk:
// 4 WMMA (dots) -> coef scale -> per-wave LDS bounce to bf16 A-fragment
// -> 4 WMMA accumulate. Cross-wave LDS reduction at the end.
// =====================================================================
struct GnnSmem {
  float wbuf[8][16 * WB_S];   // per-wave gated-dots tile, rows padded
  float red[8][16 * RED_S];   // per-wave partial outputs, rows padded
};

__global__ void gnn_kernel(const __bf16* __restrict__ e_bf,
                           const __bf16* __restrict__ eT_bf,
                           const float*  __restrict__ coef,
                           const float*  __restrict__ deg,
                           float* __restrict__ e_new)
{
  __shared__ GnnSmem S;
  const int tid = threadIdx.x, wave = tid >> 5, lane = tid & 31;
  const int g = lane >> 4, nl = lane & 15;
  const int b  = blockIdx.x >> 6;
  const int i0 = (blockIdx.x & 63) * 16;
  const __bf16* Eb  = e_bf  + (size_t)b * N_ * H_;
  const __bf16* ETb = eT_bf + (size_t)b * H_ * N_;

  // A1 fragments: rows i0..i0+15 of E (resident)
  BFrag a1[2];
  {
    const uint4v* row = reinterpret_cast<const uint4v*>(&Eb[(size_t)(i0 + nl) * H_]);
    #pragma unroll
    for (int kc = 0; kc < 2; ++kc) {
      a1[kc].q[0] = row[(kc * 32 +      g * 8) >> 3];
      a1[kc].q[1] = row[(kc * 32 + 16 + g * 8) >> 3];
    }
  }
  v8f out[4];
  #pragma unroll
  for (int ht = 0; ht < 4; ++ht) out[ht] = (v8f){0,0,0,0,0,0,0,0};

  for (int it = 0; it < N_ / 256; ++it) {
    const int j0 = it * 256 + wave * 32;
    if (it + 1 < N_ / 256)
      __builtin_prefetch(&Eb[(size_t)(j0 + 256) * H_], 0, 1);  // global_prefetch_b8

    // ---- dots tiles: D[s] = E_i (16x64) x E_j^T, j-subtile s ----
    v8f d[2];
    #pragma unroll
    for (int s = 0; s < 2; ++s) {
      const int j = j0 + s * 16 + nl;
      const uint4v* row = reinterpret_cast<const uint4v*>(&Eb[(size_t)j * H_]);
      BFrag bf0, bf1;                         // B frags over k=h
      const int q0 = (g * 8) >> 3, q1 = (32 + g * 8) >> 3;
      bf0.q[0] = row[q0]; bf0.q[1] = row[q0 + 2];
      bf1.q[0] = row[q1]; bf1.q[1] = row[q1 + 2];
      v8f acc = (v8f){0,0,0,0,0,0,0,0};
      acc = __builtin_amdgcn_wmma_f32_16x16x32_bf16(false, a1[0].v, false, bf0.v,
                                                    (short)0, acc, false, false);
      acc = __builtin_amdgcn_wmma_f32_16x16x32_bf16(false, a1[1].v, false, bf1.v,
                                                    (short)0, acc, false, false);
      d[s] = acc;
    }
    // ---- scale by coef, store W tile (16x32 f32) to per-wave LDS ----
    #pragma unroll
    for (int s = 0; s < 2; ++s) {
      #pragma unroll
      for (int v = 0; v < 8; ++v) {
        const int m = v + 8 * g;
        const float cw = coef[(size_t)(i0 + m) * N_ + j0 + s * 16 + nl];
        S.wbuf[wave][m * WB_S + s * 16 + nl] = d[s][v] * cw;
      }
    }
    // ---- rebuild as bf16 A-fragment (K = 32 j's) ----
    BFrag a2;
    {
      const float* wrow = &S.wbuf[wave][nl * WB_S];
      #pragma unroll
      for (int e2 = 0; e2 < 8; ++e2) {
        a2.v[e2]     = (__bf16)wrow[g * 8 + e2];        // elems 0..7 : k=g*8+e
        a2.v[8 + e2] = (__bf16)wrow[16 + g * 8 + e2];   // elems 8..15: k=16+g*8+e
      }
    }
    // ---- out += W x E_j  over 4 h-tiles, B from E^T (contiguous j) ----
    #pragma unroll
    for (int ht = 0; ht < 4; ++ht) {
      const int hc = ht * 16 + nl;
      const uint4v* trow = reinterpret_cast<const uint4v*>(&ETb[(size_t)hc * N_]);
      BFrag b2;
      const int qb = (j0 + g * 8) >> 3;
      b2.q[0] = trow[qb]; b2.q[1] = trow[qb + 2];
      out[ht] = __builtin_amdgcn_wmma_f32_16x16x32_bf16(false, a2.v, false, b2.v,
                                                        (short)0, out[ht], false, false);
    }
  }

  // ---- cross-wave reduction + deg normalize ----
  #pragma unroll
  for (int ht = 0; ht < 4; ++ht)
    #pragma unroll
    for (int v = 0; v < 8; ++v)
      S.red[wave][(v + 8 * g) * RED_S + ht * 16 + nl] = out[ht][v];
  __syncthreads();
  for (int cell = tid; cell < 16 * H_; cell += 256) {
    const int m = cell >> 6, h = cell & 63;
    float acc = 0.f;
    #pragma unroll
    for (int w = 0; w < 8; ++w) acc += S.red[w][m * RED_S + h];
    e_new[(size_t)b * N_ * H_ + (size_t)i0 * H_ + cell] = acc * rcp_(deg[i0 + m]);
  }
}

// =====================================================================
// Kernel 4: preds = [e_old, e_new] @ pred_w^T + pred_b
// =====================================================================
__global__ void pred_kernel(const float* __restrict__ e_old, const float* __restrict__ e_new,
                            const float* __restrict__ pw, const float* __restrict__ pb,
                            float* __restrict__ outp)
{
  __shared__ float w[2 * H_];
  const int tid = threadIdx.x;
  if (tid < 2 * H_) w[tid] = pw[tid];
  __syncthreads();
  const int bn = blockIdx.x * 256 + tid;
  float acc = pb[0];
  const float* a = &e_old[(size_t)bn * H_];
  const float* c = &e_new[(size_t)bn * H_];
  #pragma unroll 8
  for (int h = 0; h < H_; ++h) acc += a[h] * w[h];
  #pragma unroll 8
  for (int h = 0; h < H_; ++h) acc += c[h] * w[64 + h];
  outp[bn] = acc;
}

// =====================================================================
extern "C" void kernel_launch(void* const* d_in, const int* in_sizes, int n_in,
                              void* d_out, int out_size, void* d_ws, size_t ws_size,
                              hipStream_t stream)
{
  const float* x    = (const float*)d_in[0];
  const float* A    = (const float*)d_in[1];
  const float* Wih  = (const float*)d_in[2];
  const float* Whh  = (const float*)d_in[3];
  const float* bih  = (const float*)d_in[4];
  const float* bhh  = (const float*)d_in[5];
  const float* attw = (const float*)d_in[6];
  const float* attb = (const float*)d_in[7];
  const float* gw   = (const float*)d_in[8];
  const float* gb   = (const float*)d_in[9];
  const float* pw   = (const float*)d_in[10];
  const float* pb   = (const float*)d_in[11];
  float* outp = (float*)d_out;

  char* ws = (char*)d_ws;
  size_t off = 0;
  auto wsalloc = [&](size_t bytes) -> void* {
    void* p = ws + off;
    off = (off + bytes + 255) & ~(size_t)255;
    return p;
  };
  float*  e_old = (float*) wsalloc(sizeof(float)  * B_ * N_ * H_);   // 8.4 MB
  float*  e_new = (float*) wsalloc(sizeof(float)  * B_ * N_ * H_);   // 8.4 MB
  float*  coef  = (float*) wsalloc(sizeof(float)  * N_ * N_);        // 4.2 MB
  float*  deg   = (float*) wsalloc(sizeof(float)  * N_);
  __bf16* e_bf  = (__bf16*)wsalloc(sizeof(__bf16) * B_ * N_ * H_);   // 4.2 MB
  __bf16* eT_bf = (__bf16*)wsalloc(sizeof(__bf16) * B_ * N_ * H_);   // 4.2 MB

  // allow >64KB dynamic LDS for the fused LSTM kernel (WGP has 320KB)
  (void)hipFuncSetAttribute((const void*)lstm_attn_kernel,
                            hipFuncAttributeMaxDynamicSharedMemorySize,
                            (int)sizeof(LstmSmem));

  coef_kernel<<<N_, 256, 0, stream>>>(A, gw, gb, coef, deg);
  lstm_attn_kernel<<<(B_ * N_) / 16, 256, sizeof(LstmSmem), stream>>>(
      x, Wih, Whh, bih, bhh, attw, attb, e_old, e_bf, eT_bf);
  gnn_kernel<<<B_ * (N_ / 16), 256, 0, stream>>>(e_bf, eT_bf, coef, deg, e_new);
  pred_kernel<<<(B_ * N_) / 256, 256, 0, stream>>>(e_old, e_new, pw, pb, outp);
}
